// Model_23313082482696
// MI455X (gfx1250) — matrix operand order repaired and checked by reading, not compiled, not run on hardware
//
#include <hip/hip_runtime.h>
#include <hip/hip_bf16.h>

typedef __attribute__((ext_vector_type(16))) _Float16 v16h;
typedef __attribute__((ext_vector_type(8)))  _Float16 v8h;
typedef __attribute__((ext_vector_type(8)))  float    v8f;

#define G_DIM 256
#define C_DIM 512
#define D_DIM 128
#define B_DIM 1024
#define I_DIM 20000

// ---------------------------------------------------------------------------
// m1 = softmax(user_table[user_idx]) over G=256. One block per row.
// ---------------------------------------------------------------------------
__global__ void m1_kernel(const float* __restrict__ user_table,
                          const int* __restrict__ user_idx,
                          float* __restrict__ m1_out, _Float16* __restrict__ m1h) {
  __shared__ float red[256];
  const int b = blockIdx.x, t = threadIdx.x;
  const int u = user_idx[b];
  float x = user_table[(size_t)u * G_DIM + t];
  red[t] = x; __syncthreads();
  for (int s = 128; s > 0; s >>= 1) { if (t < s) red[t] = fmaxf(red[t], red[t + s]); __syncthreads(); }
  const float mx = red[0]; __syncthreads();
  const float e = __expf(x - mx);
  red[t] = e; __syncthreads();
  for (int s = 128; s > 0; s >>= 1) { if (t < s) red[t] += red[t + s]; __syncthreads(); }
  const float v = e / red[0];
  m1_out[(size_t)b * G_DIM + t] = v;
  m1h[(size_t)b * G_DIM + t] = (_Float16)v;
}

// ---------------------------------------------------------------------------
// m2 = softmax(item_table) over C=512. One block per row, 2 elems/thread.
// ---------------------------------------------------------------------------
__global__ void m2_kernel(const float* __restrict__ item_table,
                          float* __restrict__ m2_out, _Float16* __restrict__ m2h) {
  __shared__ float red[256];
  const int r = blockIdx.x, t = threadIdx.x;
  const float* row = item_table + (size_t)r * C_DIM;
  const float x0 = row[t], x1 = row[t + 256];
  red[t] = fmaxf(x0, x1); __syncthreads();
  for (int s = 128; s > 0; s >>= 1) { if (t < s) red[t] = fmaxf(red[t], red[t + s]); __syncthreads(); }
  const float mx = red[0]; __syncthreads();
  const float e0 = __expf(x0 - mx), e1 = __expf(x1 - mx);
  red[t] = e0 + e1; __syncthreads();
  for (int s = 128; s > 0; s >>= 1) { if (t < s) red[t] += red[t + s]; __syncthreads(); }
  const float inv = 1.0f / red[0];
  m2_out[(size_t)r * C_DIM + t]       = e0 * inv;
  m2_out[(size_t)r * C_DIM + t + 256] = e1 * inv;
  m2h[(size_t)r * C_DIM + t]       = (_Float16)(e0 * inv);
  m2h[(size_t)r * C_DIM + t + 256] = (_Float16)(e1 * inv);
}

// ---------------------------------------------------------------------------
// kl_gc[g,c] = 0.5*(sum ls2 - sum ls1 + sum (sig1+(mu1-mu2)^2)*exp(-ls2) - D)
// One block per g; group row cached in LDS; 2 category cols per thread.
// Also emits Eh = f16(exp(-kl_gc)) (B operand of the logB GEMM).
// ---------------------------------------------------------------------------
__global__ void klgc_kernel(const float* __restrict__ gmu, const float* __restrict__ gls,
                            const float* __restrict__ cmu, const float* __restrict__ cls,
                            float* __restrict__ klgc, _Float16* __restrict__ Eh) {
  __shared__ float mu1s[D_DIM], sig1s[D_DIM], red[256];
  const int g = blockIdx.x, t = threadIdx.x;
  float ls1v = 0.f;
  if (t < D_DIM) {
    const float m = gmu[g * D_DIM + t], l = gls[g * D_DIM + t];
    mu1s[t] = m; sig1s[t] = __expf(l); ls1v = l;
  }
  red[t] = ls1v; __syncthreads();
  for (int s = 128; s > 0; s >>= 1) { if (t < s) red[t] += red[t + s]; __syncthreads(); }
  const float s1 = red[0]; __syncthreads();
  for (int cc = 0; cc < 2; ++cc) {
    const int c = t + cc * 256;
    const float* mrow = cmu + (size_t)c * D_DIM;
    const float* lrow = cls + (size_t)c * D_DIM;
    float acc = 0.f, ls2sum = 0.f;
    for (int d = 0; d < D_DIM; ++d) {
      const float l2 = lrow[d];
      const float inv2 = __expf(-l2);
      const float dm = mu1s[d] - mrow[d];
      acc = fmaf(sig1s[d] + dm * dm, inv2, acc);
      ls2sum += l2;
    }
    const float kl = 0.5f * (ls2sum - s1 + acc - (float)D_DIM);
    klgc[(size_t)g * C_DIM + c] = kl;
    Eh[(size_t)g * C_DIM + c] = (_Float16)__expf(-kl);
  }
}

// ---------------------------------------------------------------------------
// tp = sigmoid(w * standardize_row(kl_gc) + b), stored TRANSPOSED in f16
// (tpT[c,g]) so P1 = m1 @ tp becomes an NT WMMA GEMM.
// ---------------------------------------------------------------------------
__global__ void tp_kernel(const float* __restrict__ klgc,
                          const float* __restrict__ lin_w, const float* __restrict__ lin_b,
                          _Float16* __restrict__ tpT) {
  __shared__ float red[256], red2[256];
  const int g = blockIdx.x, t = threadIdx.x;
  const float x0 = klgc[(size_t)g * C_DIM + t], x1 = klgc[(size_t)g * C_DIM + t + 256];
  red[t] = x0 + x1; red2[t] = x0 * x0 + x1 * x1; __syncthreads();
  for (int s = 128; s > 0; s >>= 1) {
    if (t < s) { red[t] += red[t + s]; red2[t] += red2[t + s]; } __syncthreads();
  }
  const float mean = red[0] * (1.0f / C_DIM);
  const float var  = (red2[0] - (float)C_DIM * mean * mean) * (1.0f / (C_DIM - 1));
  const float rstd = rsqrtf(fmaxf(var, 1e-20f));
  const float w = lin_w[0], bb = lin_b[0];
  const float z0 = (x0 - mean) * rstd, z1 = (x1 - mean) * rstd;
  tpT[(size_t)t * G_DIM + g]         = (_Float16)(1.0f / (1.0f + __expf(-(w * z0 + bb))));
  tpT[(size_t)(t + 256) * G_DIM + g] = (_Float16)(1.0f / (1.0f + __expf(-(w * z1 + bb))));
}

// ---------------------------------------------------------------------------
// NT WMMA GEMM: C(M,N) f32 = A(M,K) f16 row-major x Bt(N,K) f16 row-major ^T.
// ONE WAVE PER WORKGROUP (blockDim=32): every tile index derives from
// blockIdx (SGPR), so all control flow is scalar and EXEC stays all-ones
// around WMMA. Each wave computes a 16x64 strip (4 N-tiles): the A fragment
// is loaded once per k-step and reused by 4 v_wmma_f32_16x16x32_f16 with 4
// independent accumulator chains (covers the WMMA->WMMA RAW hazard window).
// N-edge handled by a scalar clamp: edge blocks recompute neighbor tiles and
// write identical values (branch-free, deterministic).
// Fragment k order k = hi*16 + s, same in A and B -> contiguous 16B loads per lane.
// K is compile-time (full unroll, constant offsets). MODE 1 = log epilogue.
// ---------------------------------------------------------------------------
template <int KV, int MODE>
__global__ void gemm_nt_wmma(const _Float16* __restrict__ A,
                             const _Float16* __restrict__ Bt,
                             float* __restrict__ Cf32, _Float16* __restrict__ Cf16,
                             int N) {
  const int lane = threadIdx.x & 31;
  const int nTiles = N >> 4;
  int nt0 = blockIdx.x * 4;                    // scalar (SGPR)
  if (nt0 + 4 > nTiles) nt0 = nTiles - 4;      // scalar clamp (overlap-safe)
  const int mt = blockIdx.y;

  const int hi  = lane >> 4;                   // 0: lanes 0-15, 1: lanes 16-31
  const int l15 = lane & 15;
  const _Float16* Ap = A + (size_t)(mt * 16 + l15) * KV + hi * 16;
  const _Float16* Bp = Bt + (size_t)(nt0 * 16 + l15) * KV + hi * 16;

  v8f acc0 = {}, acc1 = {}, acc2 = {}, acc3 = {};
#pragma unroll
  for (int k = 0; k < KV; k += 32) {
    // A fragment: lane holds row l15, K = hi*16 + {0..15}
    const v8h a0 = *(const v8h*)(Ap + k);
    const v8h a1 = *(const v8h*)(Ap + k + 8);
    v16h av;
#pragma unroll
    for (int i = 0; i < 8; ++i) { av[i] = a0[i]; av[i + 8] = a1[i]; }

    // B fragments: lane holds tile column l15, K = hi*16 + {0..15}
    v16h bv0, bv1, bv2, bv3;
#pragma unroll
    for (int i = 0; i < 8; ++i) {
      bv0[i] = Bp[0 * 16 * KV + k + i];  bv0[i + 8] = Bp[0 * 16 * KV + k + 8 + i];
      bv1[i] = Bp[1 * 16 * KV + k + i];  bv1[i + 8] = Bp[1 * 16 * KV + k + 8 + i];
      bv2[i] = Bp[2 * 16 * KV + k + i];  bv2[i + 8] = Bp[2 * 16 * KV + k + 8 + i];
      bv3[i] = Bp[3 * 16 * KV + k + i];  bv3[i + 8] = Bp[3 * 16 * KV + k + 8 + i];
    }
    acc0 = __builtin_amdgcn_wmma_f32_16x16x32_f16(false, av, false, bv0, (short)0, acc0, false, false);
    acc1 = __builtin_amdgcn_wmma_f32_16x16x32_f16(false, av, false, bv1, (short)0, acc1, false, false);
    acc2 = __builtin_amdgcn_wmma_f32_16x16x32_f16(false, av, false, bv2, (short)0, acc2, false, false);
    acc3 = __builtin_amdgcn_wmma_f32_16x16x32_f16(false, av, false, bv3, (short)0, acc3, false, false);

    // stream-ahead hints (global_prefetch_b8: counter-free, speculative)
    __builtin_prefetch((const void*)(Bp + 0 * 16 * KV + k + 64), 0, 3);
    __builtin_prefetch((const void*)(Bp + 1 * 16 * KV + k + 64), 0, 3);
    __builtin_prefetch((const void*)(Bp + 2 * 16 * KV + k + 64), 0, 3);
    __builtin_prefetch((const void*)(Bp + 3 * 16 * KV + k + 64), 0, 3);
  }

  // Store: lane holds C column l15; acc VGPR v -> row v (+8 for hi lanes).
  v8f accs[4] = {acc0, acc1, acc2, acc3};
#pragma unroll
  for (int j = 0; j < 4; ++j) {
    const int col = (nt0 + j) * 16 + l15;
#pragma unroll
    for (int v = 0; v < 8; ++v) {
      const int row = mt * 16 + v + hi * 8;
      float val = accs[j][v];
      if (MODE == 1) val = __logf(fmaxf(val, 1e-30f));
      if (Cf32) Cf32[(size_t)row * N + col] = val;
      if (Cf16) Cf16[(size_t)row * N + col] = (_Float16)val;
    }
  }
}

// ---------------------------------------------------------------------------
// In-place row softmax / row normalize over N columns. One block per row.
// ---------------------------------------------------------------------------
__global__ void softmax_rows(float* __restrict__ data, int N) {
  __shared__ float red[256];
  const int r = blockIdx.x, t = threadIdx.x;
  float* row = data + (size_t)r * N;
  float mx = -3.4e38f;
  for (int i = t; i < N; i += 256) mx = fmaxf(mx, row[i]);
  red[t] = mx; __syncthreads();
  for (int s = 128; s > 0; s >>= 1) { if (t < s) red[t] = fmaxf(red[t], red[t + s]); __syncthreads(); }
  mx = red[0]; __syncthreads();
  float sum = 0.f;
  for (int i = t; i < N; i += 256) sum += __expf(row[i] - mx);
  red[t] = sum; __syncthreads();
  for (int s = 128; s > 0; s >>= 1) { if (t < s) red[t] += red[t + s]; __syncthreads(); }
  const float inv = 1.0f / red[0];
  for (int i = t; i < N; i += 256) row[i] = __expf(row[i] - mx) * inv;
}

__global__ void normalize_rows(float* __restrict__ data, int N) {
  __shared__ float red[256];
  const int r = blockIdx.x, t = threadIdx.x;
  float* row = data + (size_t)r * N;
  float sum = 0.f;
  for (int i = t; i < N; i += 256) sum += row[i];
  red[t] = sum; __syncthreads();
  for (int s = 128; s > 0; s >>= 1) { if (t < s) red[t] += red[t + s]; __syncthreads(); }
  const float inv = 1.0f / red[0];
  for (int i = t; i < N; i += 256) row[i] *= inv;
}

// ---------------------------------------------------------------------------
// kl_norm scalar: 0.5*sum(mu^2 + exp(ls) - ls) - 0.5*D*(G+C). One block.
// ---------------------------------------------------------------------------
__global__ void klnorm_kernel(const float* __restrict__ gmu, const float* __restrict__ gls,
                              const float* __restrict__ cmu, const float* __restrict__ cls,
                              float* __restrict__ out) {
  __shared__ float red[256];
  const int t = threadIdx.x;
  float acc = 0.f;
  for (int i = t; i < G_DIM * D_DIM; i += 256) { const float m = gmu[i], l = gls[i]; acc += m * m + __expf(l) - l; }
  for (int i = t; i < C_DIM * D_DIM; i += 256) { const float m = cmu[i], l = cls[i]; acc += m * m + __expf(l) - l; }
  red[t] = acc; __syncthreads();
  for (int s = 128; s > 0; s >>= 1) { if (t < s) red[t] += red[t + s]; __syncthreads(); }
  if (t == 0) out[0] = 0.5f * red[0] - 0.5f * (float)D_DIM * (float)(G_DIM + C_DIM);
}

// ---------------------------------------------------------------------------
extern "C" void kernel_launch(void* const* d_in, const int* in_sizes, int n_in,
                              void* d_out, int out_size, void* d_ws, size_t ws_size,
                              hipStream_t stream) {
  const float* gmu  = (const float*)d_in[0];
  const float* gls  = (const float*)d_in[1];
  const float* cmu  = (const float*)d_in[2];
  const float* cls  = (const float*)d_in[3];
  const float* utab = (const float*)d_in[4];
  const float* itab = (const float*)d_in[5];
  const float* linw = (const float*)d_in[6];
  const float* linb = (const float*)d_in[7];
  const int*   uidx = (const int*)d_in[8];

  float* out   = (float*)d_out;
  float* mix   = out;                                   // (1024, 20000)
  float* trans = out + (size_t)B_DIM * I_DIM;           // (1024, 20000)
  float* klsum = out + 2ull * B_DIM * I_DIM;            // scalar
  float* m1out = klsum + 1;                             // (1024, 256)
  float* m2out = m1out + (size_t)B_DIM * G_DIM;         // (20000, 512)

  char* ws = (char*)d_ws;
  _Float16* m1h   = (_Float16*)(ws);                    //  0.50 MB
  _Float16* m2h   = (_Float16*)(ws + 524288);           // 20.48 MB
  float*    klgc  = (float*)   (ws + 21004288);         //  0.52 MB
  _Float16* Eh    = (_Float16*)(ws + 21528576);         //  0.26 MB
  _Float16* tpT   = (_Float16*)(ws + 21790720);         //  0.26 MB
  _Float16* P1h   = (_Float16*)(ws + 22052864);         //  1.05 MB
  _Float16* logBh = (_Float16*)(ws + 23101440);         // 10.24 MB

  m1_kernel<<<B_DIM, 256, 0, stream>>>(utab, uidx, m1out, m1h);
  m2_kernel<<<I_DIM, 256, 0, stream>>>(itab, m2out, m2h);
  klgc_kernel<<<G_DIM, 256, 0, stream>>>(gmu, gls, cmu, cls, klgc, Eh);
  tp_kernel<<<G_DIM, 256, 0, stream>>>(klgc, linw, linb, tpT);

  // One wave (32 threads) per block; each block computes a 16x64 C strip.
  // logB = log(m2 @ exp(-kl_gc)^T) : M=20000, N=256, K=512 -> f16 + log
  gemm_nt_wmma<512, 1><<<dim3(G_DIM / 64, I_DIM / 16), 32, 0, stream>>>(
      m2h, Eh, nullptr, logBh, G_DIM);
  // P1 = m1 @ tp : M=1024, N=512, K=256 -> f16
  gemm_nt_wmma<256, 0><<<dim3(C_DIM / 64, B_DIM / 16), 32, 0, stream>>>(
      m1h, tpT, nullptr, P1h, C_DIM);
  // S = m1 @ logB^T : M=1024, N=20000, K=256 -> f32 into mixture slot
  gemm_nt_wmma<256, 0><<<dim3((I_DIM / 16 + 3) / 4, B_DIM / 16), 32, 0, stream>>>(
      m1h, logBh, mix, nullptr, I_DIM);
  // mixture_prob = row_softmax(S)   (per-row constant of mix_kl cancels)
  softmax_rows<<<B_DIM, 256, 0, stream>>>(mix, I_DIM);

  // prob = P1 @ m2^T : M=1024, N=20000, K=512 -> f32 into transition slot
  gemm_nt_wmma<512, 0><<<dim3((I_DIM / 16 + 3) / 4, B_DIM / 16), 32, 0, stream>>>(
      P1h, m2h, trans, nullptr, I_DIM);
  normalize_rows<<<B_DIM, 256, 0, stream>>>(trans, I_DIM);

  klnorm_kernel<<<1, 256, 0, stream>>>(gmu, gls, cmu, cls, klsum);
}
